// SE3EquivLayer_77738908058317
// MI455X (gfx1250) — matrix operand order repaired and checked by reading, not compiled, not run on hardware
//
#include <hip/hip_runtime.h>

typedef __attribute__((ext_vector_type(16))) __bf16 v16bf;
typedef __attribute__((ext_vector_type(8)))  float  v8f;

#define N_NODES_   10000
#define N_EDGES_   160000
#define MUL_       16
#define NB_        10
#define HID_       64
#define TILE_E     16
#define NTILES     (N_EDGES_ / TILE_E)
#define THREADS    256

// folded constants
#define SCALE_S    0.011048543456f   // (1/sqrt(2*MUL)) * N/E = 0.17677669 * 0.0625
#define INV_SQRT3  0.57735026919f
#define SQRT3      1.73205080757f
#define SQRT2      1.41421356237f
#define W1_SCALE   0.31622776601f    // 1/sqrt(NUM_BASIS)
#define W2_SCALE   0.125f            // 1/sqrt(HIDDEN)
#define EMB_CONST  26.669037f        // 1.14136 * e^2 * sqrt(10)
#define RSTEP      (4.5f / 11.0f)

__global__ void se3_zero_kernel(float* __restrict__ p, int n) {
    int i = blockIdx.x * blockDim.x + threadIdx.x;
    if (i < n) p[i] = 0.0f;
}

__global__ __launch_bounds__(THREADS)
void se3_main_kernel(const float* __restrict__ x,
                     const float* __restrict__ edge_vec,
                     const float* __restrict__ w1,
                     const float* __restrict__ w2,
                     const int*   __restrict__ src,
                     const int*   __restrict__ dst,
                     float* __restrict__ out)
{
    // W2 staged once per block, pre-swizzled into WMMA B-fragment order:
    // fragment element i of lane L of (n-tile nt, k32-chunk c) is contiguous.
    __shared__ v16bf sB[64 * 2 * 32];                 // 128 KB
    __shared__ float sW1[NB_][HID_];                  // 2.5 KB
    __shared__ float sEmb[TILE_E][NB_];
    __shared__ __align__(16) __bf16 sH[TILE_E][HID_]; // A matrix (16 edges x 64 k)
    __shared__ float sX0[TILE_E][MUL_];
    __shared__ float sD [TILE_E][MUL_];
    __shared__ float sX1[TILE_E][MUL_][3];
    __shared__ float sY1[TILE_E][3];
    __shared__ float sSum[TILE_E][64];                // per-tile summand

    const int tid = threadIdx.x;

    // ---- one-time: stage & swizzle W2 into LDS as bf16 B-fragments ----
    for (int idx = tid; idx < 64 * 1024; idx += THREADS) {
        int i    = idx & 15;           // element within lane fragment
        int lane = (idx >> 4) & 31;
        int ck   = (idx >> 9) & 1;     // which K=32 chunk
        int nt   = idx >> 10;          // n-tile (p*16 + u)
        int k = ck * 32 + ((lane & 16) ? 8 : 0) + ((i & 8) ? 16 : 0) + (i & 7);
        int n = nt * 16 + (lane & 15);
        float v = w2[k * 1024 + n] * W2_SCALE;
        reinterpret_cast<__bf16*>(sB)[(((nt * 2 + ck) * 32 + lane) << 4) + i] = (__bf16)v;
    }
    for (int idx = tid; idx < NB_ * HID_; idx += THREADS)
        sW1[idx / HID_][idx % HID_] = w1[idx] * W1_SCALE;
    __syncthreads();

    const int el   = tid >> 4;     // edge-in-tile (prep phase)
    const int j    = tid & 15;     // 16 threads per edge
    const int lane = tid & 31;
    const int wave = __builtin_amdgcn_readfirstlane(tid >> 5);  // scalar branch
    const int m    = lane & 15;    // w_out column in C tile
    const int lh   = lane >> 4;    // edge-half selector in C tile

    for (int tile = blockIdx.x; tile < NTILES; tile += (int)gridDim.x) {
        const int e = tile * TILE_E + el;

        for (int z = tid; z < TILE_E * 64; z += THREADS)
            (&sSum[0][0])[z] = 0.0f;

        // ---- per-edge geometry + radial basis + features ----
        float ev0 = edge_vec[e * 3 + 0];
        float ev1 = edge_vec[e * 3 + 1];
        float ev2 = edge_vec[e * 3 + 2];
        float r   = sqrtf(ev0 * ev0 + ev1 * ev1 + ev2 * ev2);
        float s3r = SQRT3 / r;
        float y0 = ev0 * s3r, y1c = ev1 * s3r, y2 = ev2 * s3r;
        if (j == 0) { sY1[el][0] = y0; sY1[el][1] = y1c; sY1[el][2] = y2; }
        if (j < NB_) {
            float vb   = (float)(j + 1) * RSTEP;
            float diff = (r - vb) * (1.0f / RSTEP);
            float t1 = diff + 1.0f, t2 = 1.0f - diff;
            float s1 = t1 > 0.0f ? __expf(-1.0f / t1) : 0.0f;
            float s2 = t2 > 0.0f ? __expf(-1.0f / t2) : 0.0f;
            sEmb[el][j] = EMB_CONST * s1 * s2;
        }
        // gather x[src[e]]; u == j. Scale c*N/E folded into features.
        const float* xr = x + (long)src[e] * 64;
        float x0 = xr[j];
        float a0 = xr[16 + 3 * j + 0];
        float a1 = xr[16 + 3 * j + 1];
        float a2 = xr[16 + 3 * j + 2];
        sX0[el][j]    = SCALE_S * x0;
        sX1[el][j][0] = SCALE_S * a0;
        sX1[el][j][1] = SCALE_S * a1;
        sX1[el][j][2] = SCALE_S * a2;
        sD [el][j]    = SCALE_S * INV_SQRT3 * (a0 * y0 + a1 * y1c + a2 * y2);
        __syncthreads();

        // ---- radial MLP layer 1 -> h (bf16 A matrix), 4 outputs/thread ----
        #pragma unroll
        for (int kk = 0; kk < 4; kk++) {
            int k = j * 4 + kk;
            float acc = 0.0f;
            #pragma unroll
            for (int b = 0; b < NB_; b++) acc += sEmb[el][b] * sW1[b][k];
            acc = acc > 0.0f ? SQRT2 * acc : 0.0f;
            sH[el][k] = (__bf16)acc;
        }
        __syncthreads();

        // ---- WMMA: W tile = H(16x64) @ W2(64x1024), fused einsum contraction ----
        // A-fragment per lane (bf16 16x32 layout): row M = lane%16,
        // lanes<16 hold K {0..7,16..23}, lanes>=16 hold K {8..15,24..31}.
        union { uint4 q[2]; v16bf v; } A0, A1;
        {
            const __bf16* hrow = &sH[m][0];
            const int off = lh * 8;
            A0.q[0] = *reinterpret_cast<const uint4*>(hrow + off);
            A0.q[1] = *reinterpret_cast<const uint4*>(hrow + 16 + off);
            A1.q[0] = *reinterpret_cast<const uint4*>(hrow + 32 + off);
            A1.q[1] = *reinterpret_cast<const uint4*>(hrow + 48 + off);
        }
        const int p     = wave >> 1;        // path: 0=A,1=B,2=C,3=D
        const int ubase = (wave & 1) << 3;  // this wave's u range

        if (p == 0 || p == 3) {             // out0: wA.x0  and  inv_sqrt3*wD.(x1.y1)
            float (*feat)[MUL_] = (p == 0) ? sX0 : sD;
            float acc[8];
            #pragma unroll
            for (int rr = 0; rr < 8; rr++) acc[rr] = 0.0f;
            #pragma unroll
            for (int q = 0; q < 8; q++) {
                const int nt = wave * 8 + q;
                const int u  = ubase + q;
                v16bf b0 = sB[(nt * 2 + 0) * 32 + lane];
                v16bf b1 = sB[(nt * 2 + 1) * 32 + lane];
                v8f cf = {};
                cf = __builtin_amdgcn_wmma_f32_16x16x32_bf16(false, A0.v, false, b0, (short)0, cf, false, false);
                cf = __builtin_amdgcn_wmma_f32_16x16x32_bf16(false, A1.v, false, b1, (short)0, cf, false, false);
                #pragma unroll
                for (int rr = 0; rr < 8; rr++)
                    acc[rr] += cf[rr] * feat[rr + lh * 8][u];
            }
            #pragma unroll
            for (int rr = 0; rr < 8; rr++)
                atomicAdd(&sSum[rr + lh * 8][m], acc[rr]);
        } else if (p == 1) {                // out1 += (wB.x0) (x) y1
            float acc[8];
            #pragma unroll
            for (int rr = 0; rr < 8; rr++) acc[rr] = 0.0f;
            #pragma unroll
            for (int q = 0; q < 8; q++) {
                const int nt = wave * 8 + q;
                const int u  = ubase + q;
                v16bf b0 = sB[(nt * 2 + 0) * 32 + lane];
                v16bf b1 = sB[(nt * 2 + 1) * 32 + lane];
                v8f cf = {};
                cf = __builtin_amdgcn_wmma_f32_16x16x32_bf16(false, A0.v, false, b0, (short)0, cf, false, false);
                cf = __builtin_amdgcn_wmma_f32_16x16x32_bf16(false, A1.v, false, b1, (short)0, cf, false, false);
                #pragma unroll
                for (int rr = 0; rr < 8; rr++)
                    acc[rr] += cf[rr] * sX0[rr + lh * 8][u];
            }
            #pragma unroll
            for (int rr = 0; rr < 8; rr++) {
                const int e2 = rr + lh * 8;
                atomicAdd(&sSum[e2][16 + m * 3 + 0], acc[rr] * sY1[e2][0]);
                atomicAdd(&sSum[e2][16 + m * 3 + 1], acc[rr] * sY1[e2][1]);
                atomicAdd(&sSum[e2][16 + m * 3 + 2], acc[rr] * sY1[e2][2]);
            }
        } else {                            // out1 += wC . x1   (3 m-components)
            float acc[3][8];
            #pragma unroll
            for (int rr = 0; rr < 8; rr++) { acc[0][rr] = 0.0f; acc[1][rr] = 0.0f; acc[2][rr] = 0.0f; }
            #pragma unroll
            for (int q = 0; q < 8; q++) {
                const int nt = wave * 8 + q;
                const int u  = ubase + q;
                v16bf b0 = sB[(nt * 2 + 0) * 32 + lane];
                v16bf b1 = sB[(nt * 2 + 1) * 32 + lane];
                v8f cf = {};
                cf = __builtin_amdgcn_wmma_f32_16x16x32_bf16(false, A0.v, false, b0, (short)0, cf, false, false);
                cf = __builtin_amdgcn_wmma_f32_16x16x32_bf16(false, A1.v, false, b1, (short)0, cf, false, false);
                #pragma unroll
                for (int rr = 0; rr < 8; rr++) {
                    const int e2 = rr + lh * 8;
                    float cv = cf[rr];
                    acc[0][rr] += cv * sX1[e2][u][0];
                    acc[1][rr] += cv * sX1[e2][u][1];
                    acc[2][rr] += cv * sX1[e2][u][2];
                }
            }
            #pragma unroll
            for (int rr = 0; rr < 8; rr++) {
                const int e2 = rr + lh * 8;
                atomicAdd(&sSum[e2][16 + m * 3 + 0], acc[0][rr]);
                atomicAdd(&sSum[e2][16 + m * 3 + 1], acc[1][rr]);
                atomicAdd(&sSum[e2][16 + m * 3 + 2], acc[2][rr]);
            }
        }
        __syncthreads();

        // ---- scatter-add summand into out[dst] (scale already folded) ----
        #pragma unroll
        for (int z = 0; z < 4; z++) {
            int flat = z * THREADS + tid;
            int ee = flat >> 6, col = flat & 63;
            float v = sSum[ee][col];
            unsafeAtomicAdd(&out[(long)dst[tile * TILE_E + ee] * 64 + col], v);
        }
        __syncthreads();
    }
}

extern "C" void kernel_launch(void* const* d_in, const int* in_sizes, int n_in,
                              void* d_out, int out_size, void* d_ws, size_t ws_size,
                              hipStream_t stream) {
    const float* x  = (const float*)d_in[0];
    const float* ev = (const float*)d_in[1];
    const float* w1 = (const float*)d_in[2];
    const float* w2 = (const float*)d_in[3];
    const int*   sr = (const int*)d_in[4];
    const int*   ds = (const int*)d_in[5];
    float* out      = (float*)d_out;

    se3_zero_kernel<<<(out_size + 255) / 256, 256, 0, stream>>>(out, out_size);
    se3_main_kernel<<<1250, THREADS, 0, stream>>>(x, ev, w1, w2, sr, ds, out);
}
